// SelfAttentionSpareBlock_60206851555563
// MI455X (gfx1250) — matrix operand order repaired and checked by reading, not compiled, not run on hardware
//
#include <hip/hip_runtime.h>
#include <hip/hip_bf16.h>

// ---------------------------------------------------------------------------
// Self-attention with adjacency mask, MI455X (gfx1250, wave32, WMMA).
//   q = X Wq^T + bq ; k = X Wk^T + bk ; v = X Wv^T + bv
//   z = (q k^T)/sqrt(D); masked: (z*adj)==0 -> -1e4; softmax; out = relu(P v)
// All GEMMs run on v_wmma_f32_16x16x32_f16 (f16 operands, f32 accum).
// Tiles staged global->LDS with GLOBAL_LOAD_ASYNC_TO_LDS_B128 (ASYNCcnt
// path), double-buffered.
// ---------------------------------------------------------------------------

typedef _Float16 v16h __attribute__((ext_vector_type(16)));
typedef _Float16 v8h  __attribute__((ext_vector_type(8)));
typedef float    v8f  __attribute__((ext_vector_type(8)));
typedef float    v4f  __attribute__((ext_vector_type(4)));
typedef int      v4i  __attribute__((ext_vector_type(4)));

typedef __attribute__((address_space(1))) v4i* gas_v4i_ptr;   // global
typedef __attribute__((address_space(3))) v4i* lds_v4i_ptr;   // LDS (32-bit)

#define N_S   4096
#define DIM   1024
#define LDSS  40          // LDS row stride in f16 (32 + 8 pad, keeps 16B align)

#if defined(__has_builtin)
# if __has_builtin(__builtin_amdgcn_global_load_async_to_lds_b128) && \
     __has_builtin(__builtin_amdgcn_s_wait_asynccnt)
#  define USE_ASYNC_LDS 1
# endif
#endif
#ifndef USE_ASYNC_LDS
# define USE_ASYNC_LDS 0
#endif

__device__ __forceinline__ void async_tile_wait() {
#if USE_ASYNC_LDS
  __builtin_amdgcn_s_wait_asynccnt(0);
#endif
}

// ---------------------------------------------------------------------------
// f32 -> f16 cast, 8 elements / thread, 16B loads & stores
// ---------------------------------------------------------------------------
__global__ void cast_f32_f16_kernel(const float* __restrict__ src,
                                    _Float16* __restrict__ dst, int n8) {
  int i = blockIdx.x * blockDim.x + threadIdx.x;
  if (i >= n8) return;
  v4f x0 = ((const v4f*)src)[2 * i];
  v4f x1 = ((const v4f*)src)[2 * i + 1];
  v8h o;
#pragma unroll
  for (int j = 0; j < 4; ++j) { o[j] = (_Float16)x0[j]; o[4 + j] = (_Float16)x1[j]; }
  ((v8h*)dst)[i] = o;
}

__global__ void zero_f32_kernel(float* __restrict__ p, int n) {
  int i = blockIdx.x * blockDim.x + threadIdx.x;
  if (i < n) p[i] = 0.0f;
}

// ---------------------------------------------------------------------------
// Stage one 128x32 f16 tile pair into LDS. 256 threads, 2 x 16B chunks each,
// unconditional (no divergent bounds check -> no exec-mask churn).
// ---------------------------------------------------------------------------
__device__ __forceinline__ void stage_tiles(
    const _Float16* __restrict__ Ag, const _Float16* __restrict__ Bg,
    int lda, int ldb, int rowBase, int colBase, int kt,
    _Float16 (*As)[LDSS], _Float16 (*Bs)[LDSS], int tid) {
#pragma unroll
  for (int it = 0; it < 2; ++it) {
    int c   = tid + it * 256;            // 0..511, always in range
    int row = c >> 2, kc = (c & 3) * 8;
    const _Float16* ga = &Ag[(size_t)(rowBase + row) * lda + kt + kc];
    const _Float16* gb = &Bg[(size_t)(colBase + row) * ldb + kt + kc];
#if USE_ASYNC_LDS
    __builtin_amdgcn_global_load_async_to_lds_b128(
        (gas_v4i_ptr)(size_t)ga, (lds_v4i_ptr)(size_t)&As[row][kc], 0, 0);
    __builtin_amdgcn_global_load_async_to_lds_b128(
        (gas_v4i_ptr)(size_t)gb, (lds_v4i_ptr)(size_t)&Bs[row][kc], 0, 0);
#else
    *(v8h*)&As[row][kc] = *(const v8h*)ga;
    *(v8h*)&Bs[row][kc] = *(const v8h*)gb;
    __builtin_prefetch(ga + 32, 0, 1);   // gfx1250 global_prefetch, next K-tile
    __builtin_prefetch(gb + 32, 0, 1);
#endif
  }
}

// ---------------------------------------------------------------------------
// Shared 128x128-tile GEMM core. 256 threads = 8 waves (4 x 2).
// C[128,128] = A[128,kdim] * B[128,kdim]^T  (tiles staged [row][k] in LDS,
// double-buffered). Each wave owns a 32(M) x 64(N) sub-tile = 2x4 wmma tiles.
// Fragment packing follows the CDNA5 ISA VGPR layouts:
//   A (16x32 f16): lanes 0-15 row=lane, K 0-7 & 16-23 ; lanes 16-31 K 8-15 & 24-31
//   B (32x16 f16): lane%16 = N col, lanes 0-15 K 0-7 & 16-23 ; lanes 16-31 K 8-15 & 24-31
// ---------------------------------------------------------------------------
__device__ __forceinline__ void gemm_core_f16(
    const _Float16* __restrict__ Ag, const _Float16* __restrict__ Bg,
    int lda, int ldb, int kdim, int rowBase, int colBase,
    _Float16 (*As)[128][LDSS], _Float16 (*Bs)[128][LDSS], v8f acc[2][4]) {
  const int tid  = threadIdx.x;
  const int lane = tid & 31;
  const int wave = tid >> 5;
  const int wm   = wave >> 1;        // 0..3  (row group of 32)
  const int wn   = wave & 1;         // 0..1  (col group of 64)
  const int half = lane >> 4;        // 0/1
  const int cn   = lane & 15;
  const int nT   = kdim >> 5;        // K-tiles of 32

  stage_tiles(Ag, Bg, lda, ldb, rowBase, colBase, 0, As[0], Bs[0], tid);
  async_tile_wait();
  __syncthreads();

  for (int t = 0; t < nT; ++t) {
    // kick off next tile into the other buffer while we compute this one
    if (t + 1 < nT)
      stage_tiles(Ag, Bg, lda, ldb, rowBase, colBase, (t + 1) * 32,
                  As[(t + 1) & 1], Bs[(t + 1) & 1], tid);

    _Float16 (*Ac)[LDSS] = As[t & 1];
    _Float16 (*Bc)[LDSS] = Bs[t & 1];

    // ---- load fragments from LDS ----
    v16h a[2], b[4];
#pragma unroll
    for (int si = 0; si < 2; ++si) {
      int r = wm * 32 + si * 16 + cn;                  // A row = lane%16
      v8h lo = *(const v8h*)&Ac[r][half * 8];          // K 0-7  / 8-15
      v8h hi = *(const v8h*)&Ac[r][16 + half * 8];     // K 16-23 / 24-31
#pragma unroll
      for (int i = 0; i < 8; ++i) { a[si][i] = lo[i]; a[si][8 + i] = hi[i]; }
    }
#pragma unroll
    for (int ct = 0; ct < 4; ++ct) {
      int c  = wn * 64 + ct * 16 + cn;                 // B row (= output col)
      int kb = half * 8;                               // K 0-7 & 16-23 / 8-15 & 24-31
      v8h lo = *(const v8h*)&Bc[c][kb];
      v8h hi = *(const v8h*)&Bc[c][kb + 16];
#pragma unroll
      for (int i = 0; i < 8; ++i) { b[ct][i] = lo[i]; b[ct][8 + i] = hi[i]; }
    }

    // ---- 8 x v_wmma_f32_16x16x32_f16 ----
#pragma unroll
    for (int si = 0; si < 2; ++si)
#pragma unroll
      for (int ct = 0; ct < 4; ++ct)
        acc[si][ct] = __builtin_amdgcn_wmma_f32_16x16x32_f16(
            false, a[si], false, b[ct], (short)0, acc[si][ct], false, false);

    async_tile_wait();   // next tile's LDS writes landed
    __syncthreads();     // everyone done reading buffer we overwrite next
  }
}

// ---------------------------------------------------------------------------
// QKV projection: Y = Xh * Wh^T + bias.
// mode 0: row-major f16 out, scaled (Q gets 1/sqrt(D), K gets 1)
// mode 1: transposed f16 out Vt[DIM][N_S] (one b128 store per tile per lane)
// ---------------------------------------------------------------------------
__global__ __launch_bounds__(256)
void qkv_gemm_kernel(const _Float16* __restrict__ Xh,
                     const _Float16* __restrict__ Wh,
                     const float* __restrict__ bias,
                     _Float16* __restrict__ outRM,
                     _Float16* __restrict__ outT,
                     float scale, int mode) {
  __shared__ _Float16 As[2][128][LDSS];
  __shared__ _Float16 Bs[2][128][LDSS];
  v8f acc[2][4] = {};
  const int rowBase = blockIdx.y * 128;
  const int colBase = blockIdx.x * 128;
  gemm_core_f16(Xh, Wh, DIM, DIM, DIM, rowBase, colBase, As, Bs, acc);

  const int lane = threadIdx.x & 31, wave = threadIdx.x >> 5;
  const int wm = wave >> 1, wn = wave & 1, half = lane >> 4, cn = lane & 15;
  const int gr0 = rowBase + wm * 32, gc0 = colBase + wn * 64;
#pragma unroll
  for (int si = 0; si < 2; ++si)
#pragma unroll
    for (int ct = 0; ct < 4; ++ct) {
      const int gcol = gc0 + ct * 16 + cn;
      const float bv = bias[gcol];
      if (mode == 0) {
#pragma unroll
        for (int r = 0; r < 8; ++r) {
          int grow = gr0 + si * 16 + half * 8 + r;
          outRM[(size_t)grow * DIM + gcol] =
              (_Float16)((acc[si][ct][r] + bv) * scale);
        }
      } else {
        v8h pk;
#pragma unroll
        for (int r = 0; r < 8; ++r) pk[r] = (_Float16)(acc[si][ct][r] + bv);
        int rowStart = gr0 + si * 16 + half * 8;   // consecutive rows r=0..7
        *(v8h*)&outT[(size_t)gcol * N_S + rowStart] = pk;
      }
    }
}

// ---------------------------------------------------------------------------
// Scores: S = Qs * K^T (scale pre-folded into Q). Mask with adj, exp(),
// store unnormalized P (f16), atomically accumulate row denominators.
// ---------------------------------------------------------------------------
__global__ __launch_bounds__(256)
void score_kernel(const _Float16* __restrict__ Qh,
                  const _Float16* __restrict__ Kh,
                  const float* __restrict__ adj,
                  _Float16* __restrict__ P,
                  float* __restrict__ rowsum) {
  __shared__ _Float16 As[2][128][LDSS];
  __shared__ _Float16 Bs[2][128][LDSS];
  v8f acc[2][4] = {};
  const int rowBase = blockIdx.y * 128;
  const int colBase = blockIdx.x * 128;
  gemm_core_f16(Qh, Kh, DIM, DIM, DIM, rowBase, colBase, As, Bs, acc);

  const int lane = threadIdx.x & 31, wave = threadIdx.x >> 5;
  const int wm = wave >> 1, wn = wave & 1, half = lane >> 4, cn = lane & 15;
  const int gr0 = rowBase + wm * 32, gc0 = colBase + wn * 64;

  float psum[2][8];
#pragma unroll
  for (int si = 0; si < 2; ++si)
#pragma unroll
    for (int r = 0; r < 8; ++r) psum[si][r] = 0.0f;

#pragma unroll
  for (int si = 0; si < 2; ++si)
#pragma unroll
    for (int ct = 0; ct < 4; ++ct)
#pragma unroll
      for (int r = 0; r < 8; ++r) {
        int grow = gr0 + si * 16 + half * 8 + r;
        int gcol = gc0 + ct * 16 + cn;
        float s = acc[si][ct][r];
        float m = adj[(size_t)grow * N_S + gcol];
        float t = s * m;
        t = (t == 0.0f) ? -10000.0f : t;     // masked OR exact-zero score
        float p = __expf(t);                 // scores are O(1); exp is safe
        P[(size_t)grow * N_S + gcol] = (_Float16)p;
        psum[si][r] += p;
      }

  // reduce over the 16 lanes sharing each row, then global atomic add
#pragma unroll
  for (int si = 0; si < 2; ++si)
#pragma unroll
    for (int r = 0; r < 8; ++r) {
      float v = psum[si][r];
      v += __shfl_xor(v, 1, 16);
      v += __shfl_xor(v, 2, 16);
      v += __shfl_xor(v, 4, 16);
      v += __shfl_xor(v, 8, 16);
      if (cn == 0)
        atomicAdd(&rowsum[gr0 + si * 16 + half * 8 + r], v);
    }
}

// ---------------------------------------------------------------------------
// out = relu( (P * V) / rowsum ), P f16 [N,N], V via Vt f16 [D,N], out f32
// ---------------------------------------------------------------------------
__global__ __launch_bounds__(256)
void out_gemm_kernel(const _Float16* __restrict__ P,
                     const _Float16* __restrict__ Vt,
                     const float* __restrict__ rowsum,
                     float* __restrict__ out) {
  __shared__ _Float16 As[2][128][LDSS];
  __shared__ _Float16 Bs[2][128][LDSS];
  v8f acc[2][4] = {};
  const int rowBase = blockIdx.y * 128;
  const int colBase = blockIdx.x * 128;
  gemm_core_f16(P, Vt, N_S, N_S, N_S, rowBase, colBase, As, Bs, acc);

  const int lane = threadIdx.x & 31, wave = threadIdx.x >> 5;
  const int wm = wave >> 1, wn = wave & 1, half = lane >> 4, cn = lane & 15;
  const int gr0 = rowBase + wm * 32, gc0 = colBase + wn * 64;

  float inv[2][8];
#pragma unroll
  for (int si = 0; si < 2; ++si)
#pragma unroll
    for (int r = 0; r < 8; ++r)
      inv[si][r] = 1.0f / rowsum[gr0 + si * 16 + half * 8 + r];

#pragma unroll
  for (int si = 0; si < 2; ++si)
#pragma unroll
    for (int ct = 0; ct < 4; ++ct)
#pragma unroll
      for (int r = 0; r < 8; ++r) {
        int grow = gr0 + si * 16 + half * 8 + r;
        int gcol = gc0 + ct * 16 + cn;
        float v = acc[si][ct][r] * inv[si][r];
        out[(size_t)grow * DIM + gcol] = fmaxf(v, 0.0f);
      }
}

// ---------------------------------------------------------------------------
extern "C" void kernel_launch(void* const* d_in, const int* in_sizes, int n_in,
                              void* d_out, int out_size, void* d_ws, size_t ws_size,
                              hipStream_t stream) {
  const float* X   = (const float*)d_in[0];
  const float* adj = (const float*)d_in[1];
  const float* Wq  = (const float*)d_in[2];
  const float* bq  = (const float*)d_in[3];
  const float* Wk  = (const float*)d_in[4];
  const float* bk  = (const float*)d_in[5];
  const float* Wv  = (const float*)d_in[6];
  const float* bv  = (const float*)d_in[7];
  float* outp      = (float*)d_out;

  // workspace layout (bytes)
  char* ws = (char*)d_ws;
  const size_t szX  = (size_t)N_S * DIM * 2;   // 8 MiB
  const size_t szW  = (size_t)DIM * DIM * 2;   // 2 MiB
  const size_t szP  = (size_t)N_S * N_S * 2;   // 32 MiB
  _Float16* Xh  = (_Float16*)(ws);
  _Float16* Whq = (_Float16*)(ws + szX);
  _Float16* Whk = (_Float16*)(ws + szX + szW);
  _Float16* Whv = (_Float16*)(ws + szX + 2 * szW);
  _Float16* Qh  = (_Float16*)(ws + szX + 3 * szW);
  _Float16* Kh  = (_Float16*)(ws + 2 * szX + 3 * szW);
  _Float16* Vt  = (_Float16*)(ws + 3 * szX + 3 * szW);
  _Float16* Pu  = (_Float16*)(ws + 4 * szX + 3 * szW);
  float*    rs  = (float*)   (ws + 4 * szX + 3 * szW + szP);

  const float scale = 0.03125f;  // 1/sqrt(1024)

  // 1) casts to f16
  {
    int n8 = (N_S * DIM) / 8;
    cast_f32_f16_kernel<<<(n8 + 255) / 256, 256, 0, stream>>>(X, Xh, n8);
    int w8 = (DIM * DIM) / 8;
    cast_f32_f16_kernel<<<(w8 + 255) / 256, 256, 0, stream>>>(Wq, Whq, w8);
    cast_f32_f16_kernel<<<(w8 + 255) / 256, 256, 0, stream>>>(Wk, Whk, w8);
    cast_f32_f16_kernel<<<(w8 + 255) / 256, 256, 0, stream>>>(Wv, Whv, w8);
  }
  // 2) zero row denominators (ws is poisoned by harness)
  zero_f32_kernel<<<(N_S + 255) / 256, 256, 0, stream>>>(rs, N_S);

  // 3) projections
  dim3 gQ(DIM / 128, N_S / 128);   // (8,32)
  qkv_gemm_kernel<<<gQ, 256, 0, stream>>>(Xh, Whq, bq, Qh, nullptr, scale, 0);
  qkv_gemm_kernel<<<gQ, 256, 0, stream>>>(Xh, Whk, bk, Kh, nullptr, 1.0f, 0);
  qkv_gemm_kernel<<<gQ, 256, 0, stream>>>(Xh, Whv, bv, nullptr, Vt, 1.0f, 1);

  // 4) masked scores + exp + row sums
  dim3 gS(N_S / 128, N_S / 128);   // (32,32)
  score_kernel<<<gS, 256, 0, stream>>>(Qh, Kh, adj, Pu, rs);

  // 5) out = relu((P V)/rowsum)
  out_gemm_kernel<<<gQ, 256, 0, stream>>>(Pu, Vt, rs, outp);

  (void)in_sizes; (void)n_in; (void)out_size; (void)ws_size;
}
